// MultiHeadedAttention_79302276153485
// MI455X (gfx1250) — matrix operand order repaired and checked by reading, not compiled, not run on hardware
//
#include <hip/hip_runtime.h>
#include <hip/hip_bf16.h>

// ---------------------------------------------------------------------------
// MI455X (gfx1250) multi-head attention, f16 WMMA everywhere.
//   B=32 T=512 C=1024 H=16 D=64.  ~172 GFLOP vs ~10us of HBM traffic at
//   23.3 TB/s -> compute bound -> v_wmma_f32_16x16x32_f16 for all GEMMs,
//   GLOBAL_LOAD_ASYNC_TO_LDS_B128 for the non-transposing tile copies.
// ---------------------------------------------------------------------------

typedef __attribute__((ext_vector_type(16))) _Float16 v16h;
typedef __attribute__((ext_vector_type(8)))  float    v8f;
typedef __attribute__((ext_vector_type(4)))  int      v4i;

#define WMMA_F16(a, b, c) \
  __builtin_amdgcn_wmma_f32_16x16x32_f16(false, (a), false, (b), (short)0, (c), false, false)

#define B_DIM 32
#define T_DIM 512
#define C_DIM 1024
#define H_DIM 16
#define D_DIM 64
#define KSTEP 32
#define ALD   40   // LDS row stride (f16) for [M][K] tiles, multiple of 8
#define BLD   40   // LDS row stride (f16) for [N][K] tiles, multiple of 8

union Frag { v16h v; uint4 q[2]; };
union Pack8 { uint4 q; _Float16 a[8]; };

// ---- async global->LDS copy (gfx1250 GLOBAL_LOAD_ASYNC_TO_LDS_B128) -------
#if defined(__AMDGCN__) && __has_builtin(__builtin_amdgcn_global_load_async_to_lds_b128)
#define ASYNC_LDS 1
using g_v4i = __attribute__((address_space(1))) v4i*;
using l_v4i = __attribute__((address_space(3))) v4i*;
#else
#define ASYNC_LDS 0
#endif

__device__ __forceinline__ void copy16(const _Float16* g, _Float16* l) {
#if ASYNC_LDS
  __builtin_amdgcn_global_load_async_to_lds_b128((g_v4i)g, (l_v4i)l, 0, 0);
#else
  *(uint4*)l = *(const uint4*)g;
#endif
}

__device__ __forceinline__ void async_wait() {
#if ASYNC_LDS
#if __has_builtin(__builtin_amdgcn_s_wait_asynccnt)
  __builtin_amdgcn_s_wait_asynccnt(0);
#else
  asm volatile("s_wait_asynccnt 0x0" ::: "memory");
#endif
#endif
}

__device__ __forceinline__ v8f vzero8() {
  v8f z;
  for (int e = 0; e < 8; ++e) z[e] = 0.0f;
  return z;
}

// A-matrix fragment (16x32 f16) from an LDS tile stored [M][K] row-major.
// ISA layout: lanes 0-15 (M=lane) hold K{0..7,16..23}; lanes 16-31 hold
// K{8..15,24..31}.  -> two contiguous 16B LDS loads per lane.
__device__ __forceinline__ v16h lds_afrag(const _Float16* base, int ld) {
  const int lane = threadIdx.x & 31;
  const _Float16* p = base + (lane & 15) * ld + (lane >> 4) * 8;
  Frag f;
  f.q[0] = *(const uint4*)(p);
  f.q[1] = *(const uint4*)(p + 16);
  return f.v;
}

// B-matrix fragment (32x16 f16) from an LDS tile stored [N][K].
// ISA layout: lanes 0-15 (N=lane) hold K{0..7,16..23}; lanes 16-31 hold K{8..15,24..31}.
__device__ __forceinline__ v16h lds_bfrag(const _Float16* base, int ld) {
  const int lane = threadIdx.x & 31;
  const _Float16* p = base + (lane & 15) * ld + (lane >> 4) * 8;
  Frag f;
  f.q[0] = *(const uint4*)(p);
  f.q[1] = *(const uint4*)(p + 16);
  return f.v;
}

// ---------------------------------------------------------------------------
__global__ void cvt_f32_f16(const float* __restrict__ in,
                            _Float16* __restrict__ out, int n) {
  int i = (blockIdx.x * blockDim.x + threadIdx.x) * 4;
  if (i + 3 < n) {
    float4 v = *(const float4*)(in + i);
    out[i + 0] = (_Float16)v.x;
    out[i + 1] = (_Float16)v.y;
    out[i + 2] = (_Float16)v.z;
    out[i + 3] = (_Float16)v.w;
  }
}

// ---------------------------------------------------------------------------
// Q/K/V projection: out[b,h,t,d] = sum_c x[b,t,c] * w[h,c,d]   (f16 out)
// grid = (T/256, H, B), 256 threads (8 waves). 256x64 block tile; each wave
// owns two 16-row M-tiles (w and w+8) so B-fragments are reused 2x.
// ---------------------------------------------------------------------------
__global__ void __launch_bounds__(256) gemm_qkv(const _Float16* __restrict__ xh,
                                                const _Float16* __restrict__ wh,
                                                _Float16* __restrict__ outh) {
  __shared__ _Float16 As[256 * ALD];  // [M=256][K=32]
  __shared__ _Float16 Bs[64 * BLD];   // [N=64][K=32]
  const int t0 = blockIdx.x * 256;
  const int h  = blockIdx.y;
  const int b  = blockIdx.z;
  const int tid  = threadIdx.x;
  const int wave = tid >> 5, lane = tid & 31;
  const int n16 = lane & 15, hlf = lane >> 4;

  const _Float16* xrow  = xh + (size_t)(b * T_DIM + t0) * C_DIM;
  const _Float16* wbase = wh + (size_t)h * C_DIM * D_DIM;

  v8f acc0[4], acc1[4];
#pragma unroll
  for (int i = 0; i < 4; ++i) { acc0[i] = vzero8(); acc1[i] = vzero8(); }

  for (int k0 = 0; k0 < C_DIM; k0 += KSTEP) {
    __syncthreads();
    // A tile: 256x32 f16 = 1024 chunks of 8; async direct copy.
#pragma unroll
    for (int i = 0; i < 4; ++i) {
      int c = tid * 4 + i;
      int row = c >> 2, kq = c & 3;
      copy16(xrow + (size_t)row * C_DIM + k0 + kq * 8,
             As + row * ALD + kq * 8);
    }
    // B tile: w[h][k][d] -> Bs[d][k] transpose-scatter (one chunk/thread).
    {
      int k = tid >> 3, dq = tid & 7;
      const _Float16* gp = wbase + (size_t)(k0 + k) * D_DIM + dq * 8;
      Pack8 u;
      u.q = *(const uint4*)gp;
#pragma unroll
      for (int e = 0; e < 8; ++e) Bs[(dq * 8 + e) * BLD + k] = u.a[e];
      if (k0 + KSTEP < C_DIM) __builtin_prefetch(gp + KSTEP * D_DIM, 0, 1);
    }
    async_wait();
    __syncthreads();

    v16h a0 = lds_afrag(As + wave * 16 * ALD, ALD);
    v16h a1 = lds_afrag(As + (wave + 8) * 16 * ALD, ALD);
    v16h bv[4];
#pragma unroll
    for (int nt = 0; nt < 4; ++nt) bv[nt] = lds_bfrag(Bs + nt * 16 * BLD, BLD);
#pragma unroll
    for (int nt = 0; nt < 4; ++nt) {
      acc0[nt] = WMMA_F16(a0, bv[nt], acc0[nt]);
      acc1[nt] = WMMA_F16(a1, bv[nt], acc1[nt]);
    }
  }

  // C layout: element r of lane L -> (M = r + 8*(L>=16), N = L%16).
  _Float16* o0 = outh + (size_t)((b * H_DIM + h) * T_DIM + t0 + wave * 16) * D_DIM;
  _Float16* o1 = o0 + (size_t)128 * D_DIM;
#pragma unroll
  for (int nt = 0; nt < 4; ++nt)
#pragma unroll
    for (int r = 0; r < 8; ++r) {
      int m = r + 8 * hlf;
      o0[(size_t)m * D_DIM + nt * 16 + n16] = (_Float16)acc0[nt][r];
      o1[(size_t)m * D_DIM + nt * 16 + n16] = (_Float16)acc1[nt][r];
    }
}

// ---------------------------------------------------------------------------
// Flash attention per (b,h,128-row tile).  scale = 1/sqrt(C) = 1/32.
// grid = (T/128, H, B), 256 threads (8 waves), wave w owns rows w*16..+15.
// Dynamic LDS: Ks[128][72] | Vs[64][136] | Ps[8][16][136]  = 70656 B.
// ---------------------------------------------------------------------------
#define KLD 72
#define VLD 136
#define PLD 136

__global__ void __launch_bounds__(256) attn_kernel(const _Float16* __restrict__ qh,
                                                   const _Float16* __restrict__ kh,
                                                   const _Float16* __restrict__ vh,
                                                   _Float16* __restrict__ attnh) {
  extern __shared__ char smem[];
  _Float16* Ks = (_Float16*)smem;                    // [s=128][d=64]  (=[N][K] for QK^T)
  _Float16* Vs = (_Float16*)(smem + 18432);          // [d=64][s=128]  (=[N][K] for PV)
  _Float16* Ps = (_Float16*)(smem + 18432 + 17408);  // per-wave [m=16][s=128] A-layout
  const int r0 = blockIdx.x * 128;
  const int h = blockIdx.y, b = blockIdx.z;
  const int bh = b * H_DIM + h;
  const int tid = threadIdx.x, wave = tid >> 5, lane = tid & 31;
  const int n16 = lane & 15, hlf = lane >> 4;
  const float scale = 0.03125f;  // 1/sqrt(1024)

  // Q fragments straight from global (rows are wave-private): A layout.
  const _Float16* qrow =
      qh + (size_t)(bh * T_DIM + r0 + wave * 16 + n16) * D_DIM;
  v16h aq[2];
#pragma unroll
  for (int ds = 0; ds < 2; ++ds) {
    Frag f;
    f.q[0] = *(const uint4*)(qrow + ds * 32 + hlf * 8);
    f.q[1] = *(const uint4*)(qrow + ds * 32 + hlf * 8 + 16);
    aq[ds] = f.v;
  }

  v8f o[4];
#pragma unroll
  for (int i = 0; i < 4; ++i) o[i] = vzero8();
  float mrow[8], lrow[8];
#pragma unroll
  for (int r = 0; r < 8; ++r) { mrow[r] = -1e30f; lrow[r] = 0.0f; }

  _Float16* Pw = Ps + wave * 16 * PLD;

  for (int s0 = 0; s0 <= r0; s0 += 128) {
    __syncthreads();
    // K tile: async direct copy (kh is [s][d] == [N][K] for QK^T).
#pragma unroll
    for (int i = 0; i < 4; ++i) {
      int c = tid * 4 + i;
      int s = c >> 3, dq = c & 7;
      copy16(kh + (size_t)(bh * T_DIM + s0 + s) * D_DIM + dq * 8,
             Ks + s * KLD + dq * 8);
    }
    // V tile: transpose-scatter to [d][s].
#pragma unroll
    for (int i = 0; i < 4; ++i) {
      int c = tid * 4 + i;
      int s = c >> 3, dq = c & 7;
      Pack8 u;
      u.q = *(const uint4*)(vh + (size_t)(bh * T_DIM + s0 + s) * D_DIM + dq * 8);
#pragma unroll
      for (int e = 0; e < 8; ++e) Vs[(dq * 8 + e) * VLD + s] = u.a[e];
    }
    async_wait();
    __syncthreads();

    // scores: 16 rows x 128 cols per wave (8 N-tiles x 2 K-steps of 32).
    v8f sc[8];
#pragma unroll
    for (int nt = 0; nt < 8; ++nt) {
      v16h b0 = lds_bfrag(Ks + nt * 16 * KLD, KLD);
      v16h b1 = lds_bfrag(Ks + nt * 16 * KLD + 32, KLD);
      v8f z = vzero8();
      z = WMMA_F16(aq[0], b0, z);
      z = WMMA_F16(aq[1], b1, z);
      sc[nt] = z;
    }

    // online softmax; row M = r + 8*hlf spans lanes {hlf*16 .. hlf*16+15}.
#pragma unroll
    for (int r = 0; r < 8; ++r) {
      int row = r0 + wave * 16 + r + 8 * hlf;
      float mx = -1e30f;
#pragma unroll
      for (int nt = 0; nt < 8; ++nt) {
        int col = s0 + nt * 16 + n16;
        float sv = sc[nt][r] * scale;
        sv = (col <= row) ? sv : -1e30f;  // causal mask
        sc[nt][r] = sv;
        mx = fmaxf(mx, sv);
      }
      for (int off = 1; off < 16; off <<= 1)
        mx = fmaxf(mx, __shfl_xor(mx, off, 32));
      float mnew  = fmaxf(mrow[r], mx);
      float alpha = __expf(mrow[r] - mnew);
      float ls = 0.0f;
#pragma unroll
      for (int nt = 0; nt < 8; ++nt) {
        float p = __expf(sc[nt][r] - mnew);
        ls += p;
        Pw[(r + 8 * hlf) * PLD + nt * 16 + n16] = (_Float16)p;  // A-layout store
      }
      for (int off = 1; off < 16; off <<= 1) ls += __shfl_xor(ls, off, 32);
      lrow[r] = lrow[r] * alpha + ls;
      mrow[r] = mnew;
#pragma unroll
      for (int nt = 0; nt < 4; ++nt) o[nt][r] *= alpha;
    }

    // out += P(16x128) x V(128x64): 4 K-steps x 4 N-tiles.
#pragma unroll
    for (int ss = 0; ss < 4; ++ss) {
      v16h a = lds_afrag(Pw + ss * 32, PLD);
      v16h bv[4];
#pragma unroll
      for (int nt = 0; nt < 4; ++nt)
        bv[nt] = lds_bfrag(Vs + nt * 16 * VLD + ss * 32, VLD);
#pragma unroll
      for (int nt = 0; nt < 4; ++nt) o[nt] = WMMA_F16(a, bv[nt], o[nt]);
    }
  }

  // normalize + store head-concat layout [B,T,H*D] as f16.
#pragma unroll
  for (int r = 0; r < 8; ++r) {
    int row = r0 + wave * 16 + r + 8 * hlf;
    float inv = 1.0f / lrow[r];
#pragma unroll
    for (int nt = 0; nt < 4; ++nt)
      attnh[(size_t)(b * T_DIM + row) * C_DIM + h * D_DIM + nt * 16 + n16] =
          (_Float16)(o[nt][r] * inv);
  }
}

// ---------------------------------------------------------------------------
// Output projection: out[m, n] = sum_k a[m,k] * w[k,n] + bias[n]  (f32 out)
// M = B*T = 16384, N = K = 1024.  grid = (M/256, N/64); 256x64 block tile,
// two M-tiles per wave.
// ---------------------------------------------------------------------------
__global__ void __launch_bounds__(256) gemm_proj(const _Float16* __restrict__ ah,
                                                 const _Float16* __restrict__ wph,
                                                 const float* __restrict__ bias,
                                                 float* __restrict__ out) {
  __shared__ _Float16 As[256 * ALD];
  __shared__ _Float16 Bs[64 * BLD];
  const int m0  = blockIdx.x * 256;
  const int nb0 = blockIdx.y * 64;
  const int tid  = threadIdx.x;
  const int wave = tid >> 5, lane = tid & 31;
  const int n16 = lane & 15, hlf = lane >> 4;

  const _Float16* arow = ah + (size_t)m0 * C_DIM;

  v8f acc0[4], acc1[4];
#pragma unroll
  for (int i = 0; i < 4; ++i) { acc0[i] = vzero8(); acc1[i] = vzero8(); }

  for (int k0 = 0; k0 < C_DIM; k0 += KSTEP) {
    __syncthreads();
#pragma unroll
    for (int i = 0; i < 4; ++i) {
      int c = tid * 4 + i;
      int row = c >> 2, kq = c & 3;
      copy16(arow + (size_t)row * C_DIM + k0 + kq * 8,
             As + row * ALD + kq * 8);
    }
    {
      int k = tid >> 3, cq = tid & 7;
      const _Float16* gp = wph + (size_t)(k0 + k) * C_DIM + nb0 + cq * 8;
      Pack8 u;
      u.q = *(const uint4*)gp;
#pragma unroll
      for (int e = 0; e < 8; ++e) Bs[(cq * 8 + e) * BLD + k] = u.a[e];
      if (k0 + KSTEP < C_DIM) __builtin_prefetch(gp + KSTEP * C_DIM, 0, 1);
    }
    async_wait();
    __syncthreads();

    v16h a0 = lds_afrag(As + wave * 16 * ALD, ALD);
    v16h a1 = lds_afrag(As + (wave + 8) * 16 * ALD, ALD);
    v16h bv[4];
#pragma unroll
    for (int nt = 0; nt < 4; ++nt) bv[nt] = lds_bfrag(Bs + nt * 16 * BLD, BLD);
#pragma unroll
    for (int nt = 0; nt < 4; ++nt) {
      acc0[nt] = WMMA_F16(a0, bv[nt], acc0[nt]);
      acc1[nt] = WMMA_F16(a1, bv[nt], acc1[nt]);
    }
  }

#pragma unroll
  for (int nt = 0; nt < 4; ++nt)
#pragma unroll
    for (int r = 0; r < 8; ++r) {
      int n = nb0 + nt * 16 + n16;
      int m = m0 + wave * 16 + r + 8 * hlf;
      out[(size_t)m * C_DIM + n] = acc0[nt][r] + bias[n];
      out[(size_t)(m + 128) * C_DIM + n] = acc1[nt][r] + bias[n];
    }
}

// ---------------------------------------------------------------------------
extern "C" void kernel_launch(void* const* d_in, const int* in_sizes, int n_in,
                              void* d_out, int out_size, void* d_ws, size_t ws_size,
                              hipStream_t stream) {
  (void)in_sizes; (void)n_in; (void)out_size; (void)ws_size;
  const float* x  = (const float*)d_in[0];
  const float* wq = (const float*)d_in[1];
  const float* wk = (const float*)d_in[2];
  const float* wv = (const float*)d_in[3];
  const float* wp = (const float*)d_in[4];
  const float* bp = (const float*)d_in[5];
  float* out = (float*)d_out;

  const int nx = B_DIM * T_DIM * C_DIM;          // 16777216
  const int nw = H_DIM * C_DIM * D_DIM;          // 1048576
  const int np = C_DIM * C_DIM;                  // 1048576

  char* ws = (char*)d_ws;
  _Float16* xh  = (_Float16*)(ws);
  _Float16* wqh = (_Float16*)(ws + 33554432);
  _Float16* wkh = (_Float16*)(ws + 35651584);
  _Float16* wvh = (_Float16*)(ws + 37748736);
  _Float16* wph = (_Float16*)(ws + 39845888);
  _Float16* qhb = (_Float16*)(ws + 41943040);
  _Float16* khb = (_Float16*)(ws + 75497472);
  _Float16* vhb = (_Float16*)(ws + 109051904);
  _Float16* ath = (_Float16*)(ws + 142606336);   // end = 176160768 bytes

  cvt_f32_f16<<<nx / 1024, 256, 0, stream>>>(x,  xh,  nx);
  cvt_f32_f16<<<nw / 1024, 256, 0, stream>>>(wq, wqh, nw);
  cvt_f32_f16<<<nw / 1024, 256, 0, stream>>>(wk, wkh, nw);
  cvt_f32_f16<<<nw / 1024, 256, 0, stream>>>(wv, wvh, nw);
  cvt_f32_f16<<<np / 1024, 256, 0, stream>>>(wp, wph, np);

  dim3 g1(T_DIM / 256, H_DIM, B_DIM);
  gemm_qkv<<<g1, 256, 0, stream>>>(xh, wqh, qhb);
  gemm_qkv<<<g1, 256, 0, stream>>>(xh, wkh, khb);
  gemm_qkv<<<g1, 256, 0, stream>>>(xh, wvh, vhb);

  dim3 g2(T_DIM / 128, H_DIM, B_DIM);
  attn_kernel<<<g2, 256, 70656, stream>>>(qhb, khb, vhb, ath);

  dim3 g3((B_DIM * T_DIM) / 256, C_DIM / 64, 1);
  gemm_proj<<<g3, 256, 0, stream>>>(ath, wph, bp, out);
}
